// CosAttention_52845277610534
// MI455X (gfx1250) — matrix operand order repaired and checked by reading, not compiled, not run on hardware
//
#include <hip/hip_runtime.h>
#include <hip/hip_bf16.h>

typedef __attribute__((ext_vector_type(16))) _Float16 v16h;
typedef __attribute__((ext_vector_type(8)))  _Float16 v8h;
typedef __attribute__((ext_vector_type(8)))  float    v8f;
typedef __attribute__((ext_vector_type(4)))  float    f4;

#define B_   8
#define N_   1568
#define C_   768
#define H_   12
#define D_   64
#define C3_  2304
#define BN_  (B_ * N_)     // 12544
#define BH_  (B_ * H_)     // 96
#define QT_  (N_ / 16)     // 98 query tiles per (b,h)
#define MT_  (BN_ / 16)    // 784 M tiles for the qkv gemm
#define NT_  (C3_ / 64)    // 36 N strips (64 wide) for the qkv gemm

static __device__ __forceinline__ v8f wmma_f16(v16h a, v16h b, v8f c) {
  // D = A(16x32 f16) * B(32x16 f16) + C(16x16 f32)
  return __builtin_amdgcn_wmma_f32_16x16x32_f16(false, a, false, b, (short)0, c,
                                                false, false);
}

static __device__ __forceinline__ v16h cat8(v8h lo, v8h hi) {
  v16h r;
#pragma unroll
  for (int e = 0; e < 8; ++e) { r[e] = lo[e]; r[e + 8] = hi[e]; }
  return r;
}

// ---------------------------------------------------------------------------
// Kernel 0: one-shot f32 -> f16 conversion (x and W), 8 elements per thread.
// ---------------------------------------------------------------------------
__global__ __launch_bounds__(256) void to_half(const float* __restrict__ src,
                                               _Float16* __restrict__ dst) {
  const size_t i = ((size_t)blockIdx.x * blockDim.x + threadIdx.x) * 8;
  f4 a = *(const f4*)(src + i);
  f4 b = *(const f4*)(src + i + 4);
  v8h o;
  o[0] = (_Float16)a.x; o[1] = (_Float16)a.y;
  o[2] = (_Float16)a.z; o[3] = (_Float16)a.w;
  o[4] = (_Float16)b.x; o[5] = (_Float16)b.y;
  o[6] = (_Float16)b.z; o[7] = (_Float16)b.w;
  *(v8h*)(dst + i) = o;
}

// ---------------------------------------------------------------------------
// Kernel 1: qkv = x @ W^T + b.  xh:[BN_,C_] f16, wh:[C3_,C_] f16 -> qkv f32.
// One wave computes a 16(M) x 64(N) strip. Software-pipelined: compute WMMAs
// from registers, then issue next k-step's loads into the same registers so
// they fly while the WMMAs execute.
// ---------------------------------------------------------------------------
__global__ __launch_bounds__(256) void qkv_gemm(const _Float16* __restrict__ xh,
                                                const _Float16* __restrict__ wh,
                                                const float* __restrict__ bias,
                                                float* __restrict__ qkv) {
  const int lane = threadIdx.x & 31;
  const int wave = blockIdx.x * (blockDim.x >> 5) + (threadIdx.x >> 5);
  const int grp = lane >> 4;   // 16-lane half
  const int l15 = lane & 15;

  const int m0 = (wave / NT_) * 16;
  const int n0 = (wave % NT_) * 64;

  v8f zero = {};
  v8f acc[4];
#pragma unroll
  for (int j = 0; j < 4; ++j) acc[j] = zero;

  // A fragment: row = m0 + l15, halves h -> K = kk + 16g + h
  const _Float16* xr = xh + (size_t)(m0 + l15) * C_ + 16 * grp;
  // B fragments: col = n0+j*16+l15, halves h -> K = kk + 16g + h
  const _Float16* wr0 = wh + (size_t)(n0 + l15) * C_ + 16 * grp;

  // prologue: fragments for kk = 0
  v8h a0 = *(const v8h*)(xr);
  v8h a1 = *(const v8h*)(xr + 8);
  v8h bb[4][2];
#pragma unroll
  for (int j = 0; j < 4; ++j) {
    const v8h* wp = (const v8h*)(wr0 + (size_t)j * 16 * C_);
    bb[j][0] = wp[0]; bb[j][1] = wp[1];
  }

  for (int kk = 0; kk < C_; kk += 32) {
    // compute from current registers
    v16h a = cat8(a0, a1);
#pragma unroll
    for (int j = 0; j < 4; ++j)
      acc[j] = wmma_f16(a, cat8(bb[j][0], bb[j][1]), acc[j]);

    // issue next k-step's loads (wrap on last iteration; uniform control flow)
    int kn = kk + 32;
    if (kn >= C_) kn = 0;
    a0 = *(const v8h*)(xr + kn);
    a1 = *(const v8h*)(xr + kn + 8);
#pragma unroll
    for (int j = 0; j < 4; ++j) {
      const v8h* wp = (const v8h*)(wr0 + (size_t)j * 16 * C_ + kn);
      bb[j][0] = wp[0]; bb[j][1] = wp[1];
    }
  }

  // ---- epilogue: bias add + store (C/D layout: reg r -> row r+8*grp, col l15)
#pragma unroll
  for (int j = 0; j < 4; ++j) {
    const int col = n0 + j * 16 + l15;
    const float bv = bias[col];
#pragma unroll
    for (int r = 0; r < 8; ++r) {
      const int row = m0 + r + 8 * grp;
      qkv[(size_t)row * C3_ + col] = acc[j][r] + bv;
    }
  }
}

// ---------------------------------------------------------------------------
// Kernel 2: L2-normalize q,k over D; pack f16 q,k as [BH_,N_,D_] and v
// transposed as [BH_,D_,N_].  One wave per (b,n,h): lane covers d and d+32.
// ---------------------------------------------------------------------------
__global__ __launch_bounds__(384) void norm_pack(const float* __restrict__ qkv,
                                                 _Float16* __restrict__ qh,
                                                 _Float16* __restrict__ kh,
                                                 _Float16* __restrict__ vt) {
  const int bn = blockIdx.x;          // 0..BN_-1
  const int h = threadIdx.x >> 5;     // 0..11
  const int lane = threadIdx.x & 31;
  const int b = bn / N_, n = bn % N_;
  const float* row = qkv + (size_t)bn * C3_ + h * D_;
  const size_t obase = ((size_t)(b * H_ + h) * N_ + n) * D_;

  // q
  float q0 = row[lane], q1 = row[lane + 32];
  float s = q0 * q0 + q1 * q1;
#pragma unroll
  for (int m = 1; m < 32; m <<= 1) s += __shfl_xor(s, m, 32);
  float inv = 1.0f / fmaxf(sqrtf(s), 1e-12f);
  qh[obase + lane]      = (_Float16)(q0 * inv);
  qh[obase + lane + 32] = (_Float16)(q1 * inv);

  // k
  float k0 = row[C_ + lane], k1 = row[C_ + lane + 32];
  s = k0 * k0 + k1 * k1;
#pragma unroll
  for (int m = 1; m < 32; m <<= 1) s += __shfl_xor(s, m, 32);
  inv = 1.0f / fmaxf(sqrtf(s), 1e-12f);
  kh[obase + lane]      = (_Float16)(k0 * inv);
  kh[obase + lane + 32] = (_Float16)(k1 * inv);

  // v (transposed: [BH_, D_, N_])
  float v0 = row[2 * C_ + lane], v1 = row[2 * C_ + lane + 32];
  const size_t vbase = (size_t)(b * H_ + h) * D_ * N_ + n;
  vt[vbase + (size_t)lane * N_]        = (_Float16)v0;
  vt[vbase + (size_t)(lane + 32) * N_] = (_Float16)v1;
}

// ---------------------------------------------------------------------------
// Kernel 3: flash attention. One wave per (b,h, 16-query tile); online softmax
// over keys in steps of 32. Pipelined: next K tiles and current V tiles are
// loaded right after the S WMMAs so the softmax VALU block hides their
// latency. P re-layout via wave-private LDS tile.
// ---------------------------------------------------------------------------
__global__ __launch_bounds__(256) void attn_flash(const _Float16* __restrict__ qh,
                                                  const _Float16* __restrict__ kh,
                                                  const _Float16* __restrict__ vt,
                                                  float* __restrict__ out) {
  __shared__ _Float16 smem[8 * 16 * 32];   // 8 waves * 16x32 half tile = 8 KB
  const int lane = threadIdx.x & 31;
  const int wid = threadIdx.x >> 5;
  const int job = blockIdx.x * 8 + wid;    // 0..9407
  const int qt = job % QT_;
  const int bh = job / QT_;                // 0..95
  const int n0 = qt * 16;
  const int grp = lane >> 4;
  const int l15 = lane & 15;

  const _Float16* Q = qh + (size_t)bh * N_ * D_;
  const _Float16* K = kh + (size_t)bh * N_ * D_;
  const _Float16* V = vt + (size_t)bh * D_ * N_;
  _Float16* pls = smem + wid * (16 * 32);

  // Q A-fragments for the two 32-wide d-chunks (held for the whole key loop)
  v16h qa[2];
#pragma unroll
  for (int s2 = 0; s2 < 2; ++s2) {
    const v8h* p = (const v8h*)(Q + (size_t)(n0 + l15) * D_ + s2 * 32 + 16 * grp);
    qa[s2] = cat8(p[0], p[1]);  // K runs base..+15
  }

  v8f zero = {};
  v8f acc[4];
#pragma unroll
  for (int j = 0; j < 4; ++j) acc[j] = zero;
  float mrow[8], lrow[8];
#pragma unroll
  for (int r = 0; r < 8; ++r) { mrow[r] = -__builtin_inff(); lrow[r] = 0.0f; }

  // prologue: K fragments for kt = 0
  v8h kc[2][4];
#pragma unroll
  for (int t = 0; t < 2; ++t) {
    const int key = t * 16 + l15;
    const v8h* kp0 = (const v8h*)(K + (size_t)key * D_ + 16 * grp);
    const v8h* kp1 = (const v8h*)(K + (size_t)key * D_ + 32 + 16 * grp);
    kc[t][0] = kp0[0]; kc[t][1] = kp0[1];
    kc[t][2] = kp1[0]; kc[t][3] = kp1[1];
  }

  for (int kt = 0; kt < N_; kt += 32) {
    // ---- S = q @ k^T for two 16-key tiles (from pre-loaded registers)
    v8f S[2];
#pragma unroll
    for (int t = 0; t < 2; ++t) {
      v8f z = zero;
      z = wmma_f16(qa[0], cat8(kc[t][0], kc[t][1]), z);
      S[t] = wmma_f16(qa[1], cat8(kc[t][2], kc[t][3]), z);
    }

    // ---- issue next iteration's K loads (wrap keeps control flow uniform)
    int ktn = kt + 32;
    if (ktn >= N_) ktn = 0;
#pragma unroll
    for (int t = 0; t < 2; ++t) {
      const int key = ktn + t * 16 + l15;
      const v8h* kp0 = (const v8h*)(K + (size_t)key * D_ + 16 * grp);
      const v8h* kp1 = (const v8h*)(K + (size_t)key * D_ + 32 + 16 * grp);
      kc[t][0] = kp0[0]; kc[t][1] = kp0[1];
      kc[t][2] = kp1[0]; kc[t][3] = kp1[1];
    }
    // ---- issue this iteration's V loads (consumed after softmax)
    v8h vc[4][2];
#pragma unroll
    for (int j = 0; j < 4; ++j) {
      const v8h* vp = (const v8h*)(V + (size_t)(j * 16 + l15) * N_ + kt + 16 * grp);
      vc[j][0] = vp[0]; vc[j][1] = vp[1];
    }

    // ---- online softmax update (row r lives across a 16-lane half)
#pragma unroll
    for (int r = 0; r < 8; ++r) {
      float mx = fmaxf(S[0][r], S[1][r]);
#pragma unroll
      for (int msk = 1; msk < 16; msk <<= 1) mx = fmaxf(mx, __shfl_xor(mx, msk, 32));
      const float newm = fmaxf(mrow[r], mx);
      const float sc = __expf(mrow[r] - newm);
      mrow[r] = newm;
      lrow[r] *= sc;
      acc[0][r] *= sc; acc[1][r] *= sc; acc[2][r] *= sc; acc[3][r] *= sc;
      const float p0 = __expf(S[0][r] - newm);
      const float p1 = __expf(S[1][r] - newm);
      S[0][r] = p0; S[1][r] = p1;
      float ps = p0 + p1;
#pragma unroll
      for (int msk = 1; msk < 16; msk <<= 1) ps += __shfl_xor(ps, msk, 32);
      lrow[r] += ps;
    }

    // ---- C-layout -> A-layout for P via wave-private LDS tile (16 x 32 f16)
#pragma unroll
    for (int t = 0; t < 2; ++t)
#pragma unroll
      for (int r = 0; r < 8; ++r)
        pls[(r + 8 * grp) * 32 + t * 16 + l15] = (_Float16)S[t][r];

    v16h pa;
    {
      const v8h* pp = (const v8h*)(pls + l15 * 32 + 16 * grp);
      pa = cat8(pp[0], pp[1]);  // cols 16g..16g+15
    }

    // ---- O += P @ V  (V fragments already resident)
#pragma unroll
    for (int j = 0; j < 4; ++j)
      acc[j] = wmma_f16(pa, cat8(vc[j][0], vc[j][1]), acc[j]);
  }

  // ---- normalize and store to out[b, n, h*64 + d] (f32)
  const int b = bh / H_, h = bh % H_;
#pragma unroll
  for (int r = 0; r < 8; ++r) {
    const float inv = 1.0f / lrow[r];
    const int n = n0 + r + 8 * grp;
    float* orow = out + ((size_t)(b * N_ + n)) * C_ + h * D_;
#pragma unroll
    for (int j = 0; j < 4; ++j) orow[j * 16 + l15] = acc[j][r] * inv;
  }
}

// ---------------------------------------------------------------------------
extern "C" void kernel_launch(void* const* d_in, const int* in_sizes, int n_in,
                              void* d_out, int out_size, void* d_ws, size_t ws_size,
                              hipStream_t stream) {
  (void)in_sizes; (void)n_in; (void)out_size; (void)ws_size;
  const float* x    = (const float*)d_in[0];   // [B,N,C]
  const float* W    = (const float*)d_in[1];   // [3C,C]
  const float* bias = (const float*)d_in[2];   // [3C]
  float* out = (float*)d_out;                  // [B,N,C]

  char* ws = (char*)d_ws;
  size_t off = 0;
  float* qkv = (float*)(ws + off);  off += (size_t)BN_ * C3_ * sizeof(float);
  const size_t head_elems = (size_t)BH_ * N_ * D_;
  _Float16* qh = (_Float16*)(ws + off); off += head_elems * sizeof(_Float16);
  _Float16* kh = (_Float16*)(ws + off); off += head_elems * sizeof(_Float16);
  _Float16* vt = (_Float16*)(ws + off); off += head_elems * sizeof(_Float16);
  _Float16* xh = (_Float16*)(ws + off); off += (size_t)BN_ * C_ * sizeof(_Float16);
  _Float16* wh = (_Float16*)(ws + off); off += (size_t)C3_ * C_ * sizeof(_Float16);

  // one-shot f32 -> f16 operand conversion (hoisted out of the GEMM hot loop)
  to_half<<<(BN_ * C_) / (8 * 256), 256, 0, stream>>>(x, xh);   // 4704 blocks
  to_half<<<(C3_ * C_) / (8 * 256), 256, 0, stream>>>(W, wh);   // 864 blocks

  // 784*36 = 28224 wave tiles, 8 waves/block -> 3528 blocks
  qkv_gemm<<<(MT_ * NT_) / 8, 256, 0, stream>>>(xh, wh, bias, qkv);
  // one block per (b,n), 12 waves (one per head)
  norm_pack<<<BN_, 384, 0, stream>>>(qkv, qh, kh, vt);
  // 96*98 = 9408 wave jobs, 8 waves/block -> 1176 blocks
  attn_flash<<<(BH_ * QT_) / 8, 256, 0, stream>>>(qh, kh, vt, out);
}
